// SlotAttention_87479893885285
// MI455X (gfx1250) — matrix operand order repaired and checked by reading, not compiled, not run on hardware
//
#include <hip/hip_runtime.h>
#include <hip/hip_bf16.h>
#include <math.h>

typedef __bf16 bf16_t;
typedef __attribute__((ext_vector_type(16))) __bf16 v16bf;
typedef __attribute__((ext_vector_type(8)))  __bf16 v8bf;
typedef __attribute__((ext_vector_type(8)))  float  v8f;

#define B_   32
#define N_   4096
#define FEAT_ 768
#define DIM_ 256
#define H_   4
#define DH_  64
#define KSLOT_ 16
#define ITERS_ 3
#define EPS_ 1e-8f
#define SCALE_ 0.125f   /* DH^-0.5 = 64^-0.5 */

// ---------------- workspace layout (bytes) ----------------
#define OFF_WK      0ull
#define OFF_WV      (OFF_WK + 196608ull*2)                  // to_k_w bf16
#define OFF_KB      (OFF_WV + 196608ull*2)                  // to_v_w bf16
#define OFF_VT      (OFF_KB + 33554432ull*2)                // k bf16 [b,j,256]
#define OFF_QB      (OFF_VT + 33554432ull*2)                // vT bf16 [b,256,4096]
#define OFF_ATTN    (OFF_QB + 131072ull*2)                  // q bf16 [b*16,256]
#define OFF_SLOTS   (OFF_ATTN + 8388608ull*2)               // attn bf16 [b,64,4096]
#define OFF_UPD     (OFF_SLOTS + 131072ull*4)               // slots f32
#define OFF_ROWSUM  (OFF_UPD + 131072ull*4)                 // updates f32
#define OFF_COLSUM  (OFF_ROWSUM + 2048ull*4)                // rowsum f32 [b*64]
// colsum f32 [b*256] ends at OFF_COLSUM + 32768

static __device__ __forceinline__ v16bf make_afrag(const bf16_t* p) {
    v8bf lo = *(const v8bf*)(p);
    v8bf hi = *(const v8bf*)(p + 8);
    return __builtin_shufflevector(lo, hi, 0,1,2,3,4,5,6,7,8,9,10,11,12,13,14,15);
}

// ---------------------------------------------------------------------------
// 0) convert projection weights to bf16
__global__ void convert_w_kernel(const float* __restrict__ wk_f,
                                 const float* __restrict__ wv_f,
                                 bf16_t* __restrict__ wk, bf16_t* __restrict__ wv) {
    int i = blockIdx.x * blockDim.x + threadIdx.x;
    if (i < 196608) {
        wk[i] = (bf16_t)wk_f[i];
        wv[i] = (bf16_t)wv_f[i];
    }
}

// 1) slots <- conditioning
__global__ void init_slots_kernel(const float* __restrict__ cond, float* __restrict__ slots) {
    int i = blockIdx.x * blockDim.x + threadIdx.x;
    if (i < B_ * KSLOT_ * DIM_) slots[i] = cond[i];
}

// ---------------------------------------------------------------------------
// 2) LayerNorm(inputs) + K/V projection via bf16 WMMA.
//    32 rows per block; k stored [b,j,256] bf16; v stored transposed [b,d,j] bf16.
__global__ __launch_bounds__(256) void ln_kv_kernel(
    const float* __restrict__ x, const float* __restrict__ g, const float* __restrict__ bvec,
    const bf16_t* __restrict__ wk, const bf16_t* __restrict__ wv,
    bf16_t* __restrict__ kout, bf16_t* __restrict__ vT)
{
    __shared__ __align__(32) bf16_t xs[32 * 776];   // padded stride (97*16B) -> no bank conflicts
    __shared__ float gS[FEAT_], bS[FEAT_];

    const int tid = threadIdx.x;
    for (int i = tid; i < FEAT_; i += 256) { gS[i] = g[i]; bS[i] = bvec[i]; }

    const long row_base = (long)blockIdx.x * 32;
    const int r  = tid >> 3;      // 0..31 row in tile
    const int qt = tid & 7;       // 8 threads per row
    const float* xrow = x + (row_base + r) * FEAT_;
    const int c0 = qt * 96;

    float s = 0.f, s2 = 0.f;
    for (int c = 0; c < 96; ++c) { float v = xrow[c0 + c]; s += v; s2 += v * v; }
    s  += __shfl_xor(s, 1, 32);  s2 += __shfl_xor(s2, 1, 32);
    s  += __shfl_xor(s, 2, 32);  s2 += __shfl_xor(s2, 2, 32);
    s  += __shfl_xor(s, 4, 32);  s2 += __shfl_xor(s2, 4, 32);
    const float mean = s * (1.0f / FEAT_);
    const float var  = s2 * (1.0f / FEAT_) - mean * mean;
    const float rstd = rsqrtf(var + 1e-5f);
    __syncthreads();   // gS/bS ready
    for (int c = 0; c < 96; ++c) {
        float v = xrow[c0 + c];
        xs[r * 776 + c0 + c] = (bf16_t)((v - mean) * rstd * gS[c0 + c] + bS[c0 + c]);
    }
    __syncthreads();

    // GEMM: [32,768] x [768,512] (k cols 0..255, v cols 256..511)
    const int wvid  = tid >> 5;
    const int lane  = tid & 31;
    const int l15   = lane & 15;
    const int lhalf = lane >> 4;

    for (int st = 0; st < 8; ++st) {
        const int tI  = wvid + 8 * st;           // 0..63
        const int mt  = tI & 1;                  // 2 row tiles
        const int c0g = (tI >> 1) * 16;          // 0..496
        const bf16_t* W  = (c0g < 256) ? wk : wv;
        const int col    = (c0g & 255) + l15;
        const bf16_t* wrow  = W + (size_t)col * FEAT_ + lhalf * 16;
        const bf16_t* arow0 = &xs[(mt * 16 + l15) * 776 + lhalf * 16];

        v8f acc = {};
        #pragma unroll
        for (int kk = 0; kk < 24; ++kk) {
            v16bf a = make_afrag(arow0 + kk * 32);
            v16bf b = *(const v16bf*)(wrow + kk * 32);
            acc = __builtin_amdgcn_wmma_f32_16x16x32_bf16(false, a, false, b,
                                                          (short)0, acc, false, false);
        }
        #pragma unroll
        for (int i = 0; i < 8; ++i) {
            const int  m   = mt * 16 + lhalf * 8 + i;
            const long rgo = row_base + m;
            const bf16_t val = (bf16_t)acc[i];
            if (c0g < 256) {
                kout[rgo * 256 + col] = val;
            } else {
                const long bb = rgo >> 12, j = rgo & 4095;
                vT[((bb * 256) + col) * 4096 + j] = val;
            }
        }
    }
}

// 3) colsum_v[b,d] = sum_j v[b,j,d]   (from vT, contiguous in j)
__global__ __launch_bounds__(256) void colsum_kernel(const bf16_t* __restrict__ vT,
                                                     float* __restrict__ colsum) {
    __shared__ float red[256];
    const int bd = blockIdx.x;            // 0..8191
    const bf16_t* rowp = vT + (size_t)bd * 4096;
    float s = 0.f;
    for (int j = threadIdx.x; j < 4096; j += 256) s += (float)rowp[j];
    red[threadIdx.x] = s; __syncthreads();
    for (int st = 128; st > 0; st >>= 1) {
        if (threadIdx.x < st) red[threadIdx.x] += red[threadIdx.x + st];
        __syncthreads();
    }
    if (threadIdx.x == 0) colsum[bd] = red[0];
}

__global__ void zero_kernel(float* __restrict__ p, int n) {
    int i = blockIdx.x * blockDim.x + threadIdx.x;
    if (i < n) p[i] = 0.f;
}

// ---------------------------------------------------------------------------
// 4) q = LN(slots) @ Wq^T  -> bf16  (one row per block)
__global__ __launch_bounds__(256) void qproj_kernel(
    const float* __restrict__ slots, const float* __restrict__ g, const float* __restrict__ bv,
    const float* __restrict__ wq, bf16_t* __restrict__ qout)
{
    __shared__ float sv[DIM_];
    __shared__ float red[256];
    const int row = blockIdx.x, t = threadIdx.x;
    const float xv = slots[(size_t)row * DIM_ + t];

    red[t] = xv; __syncthreads();
    for (int st = 128; st > 0; st >>= 1) { if (t < st) red[t] += red[t + st]; __syncthreads(); }
    const float mean = red[0] * (1.0f / DIM_); __syncthreads();
    red[t] = xv * xv; __syncthreads();
    for (int st = 128; st > 0; st >>= 1) { if (t < st) red[t] += red[t + st]; __syncthreads(); }
    const float var = red[0] * (1.0f / DIM_) - mean * mean;
    const float rstd = rsqrtf(var + 1e-5f);
    sv[t] = (xv - mean) * rstd * g[t] + bv[t];
    __syncthreads();

    const float* wrow = wq + (size_t)t * DIM_;
    float acc = 0.f;
    for (int f = 0; f < DIM_; ++f) acc += sv[f] * wrow[f];
    qout[(size_t)row * DIM_ + t] = (bf16_t)acc;
}

// ---------------------------------------------------------------------------
// 5) dots (WMMA) + joint softmax over slots*heads + rowsum + (last iter) attn_out
//    One block = one (b, 64-column j-tile); computes all 64 (slot,head) rows.
__global__ __launch_bounds__(256) void dots_softmax_kernel(
    const bf16_t* __restrict__ qb, const bf16_t* __restrict__ kb,
    bf16_t* __restrict__ attnW, float* __restrict__ rowsum,
    float* __restrict__ attn_out, int write_attn_out)
{
    __shared__ __align__(32) bf16_t qs[KSLOT_ * 256];
    __shared__ float Dt[64][68];

    const int b  = blockIdx.x >> 6;
    const int j0 = (blockIdx.x & 63) * 64;
    const int t  = threadIdx.x;

    { const bf16_t* src = qb + (size_t)b * KSLOT_ * 256;
      for (int i = t; i < KSLOT_ * 256; i += 256) qs[i] = src[i]; }
    __syncthreads();

    const int wvid = t >> 5, lane = t & 31, l15 = lane & 15, lhalf = lane >> 4;
    for (int st = 0; st < 2; ++st) {
        const int tI = wvid + 8 * st;   // 0..15
        const int h  = tI >> 2, jt = tI & 3;
        const bf16_t* arow = &qs[l15 * 256 + h * DH_ + lhalf * 16];
        const bf16_t* brow = kb + (((size_t)b * N_) + (j0 + jt * 16 + l15)) * 256
                                + h * DH_ + lhalf * 16;
        v8f acc = {};
        #pragma unroll
        for (int kk = 0; kk < 2; ++kk) {
            v16bf a  = make_afrag(arow + kk * 32);
            v16bf bf = *(const v16bf*)(brow + kk * 32);
            acc = __builtin_amdgcn_wmma_f32_16x16x32_bf16(false, a, false, bf,
                                                          (short)0, acc, false, false);
        }
        #pragma unroll
        for (int i = 0; i < 8; ++i) {
            const int m = lhalf * 8 + i;            // slot index
            Dt[h * 16 + m][jt * 16 + l15] = acc[i] * SCALE_;
        }
    }
    __syncthreads();

    // softmax over the 64 (slot,head) rows, per column (4 threads / column)
    const int cl = t >> 2, qq = t & 3;
    float mx = -1e30f;
    for (int r = qq * 16; r < qq * 16 + 16; ++r) mx = fmaxf(mx, Dt[r][cl]);
    mx = fmaxf(mx, __shfl_xor(mx, 1, 32));
    mx = fmaxf(mx, __shfl_xor(mx, 2, 32));
    float se = 0.f;
    for (int r = qq * 16; r < qq * 16 + 16; ++r) se += __expf(Dt[r][cl] - mx);
    se += __shfl_xor(se, 1, 32);
    se += __shfl_xor(se, 2, 32);
    const float inv = 1.0f / se;
    for (int r = qq * 16; r < qq * 16 + 16; ++r) {
        const float a = __expf(Dt[r][cl] - mx) * inv;
        Dt[r][cl] = a;
        attnW[((size_t)b * 64 + r) * N_ + j0 + cl] = (bf16_t)a;
    }
    __syncthreads();

    if (t < 64) {                       // row sums (over j) for the EPS renorm
        float s = 0.f;
        for (int c = 0; c < 64; ++c) s += Dt[t][c];
        atomicAdd(&rowsum[b * 64 + t], s);
    }
    if (write_attn_out) {               // attn_out = mean over heads (last iter only)
        for (int rep = 0; rep < 4; ++rep) {
            const int idx = t + rep * 256;       // 0..1023
            const int i = idx >> 6, c = idx & 63;
            const float v = 0.25f * (Dt[i][c] + Dt[16 + i][c] + Dt[32 + i][c] + Dt[48 + i][c]);
            attn_out[((size_t)b * KSLOT_ + i) * N_ + j0 + c] = v;
        }
    }
}

// ---------------------------------------------------------------------------
// 6) updates = (attn @ v + EPS*colsum_v) / (rowsum + N*EPS)   (WMMA over K=4096)
__global__ __launch_bounds__(64) void updates_kernel(
    const bf16_t* __restrict__ attnW, const bf16_t* __restrict__ vT,
    const float* __restrict__ rowsum, const float* __restrict__ colsum,
    float* __restrict__ updates)
{
    const int b = blockIdx.x >> 2, h = blockIdx.x & 3;
    const int t = threadIdx.x;
    const int wvid = t >> 5, lane = t & 31, l15 = lane & 15, lhalf = lane >> 4;

    for (int st = 0; st < 2; ++st) {
        const int dt = wvid + 2 * st;   // 0..3 (16-col d tiles of this head)
        const bf16_t* arow = attnW + ((size_t)b * 64 + h * 16 + l15) * N_ + lhalf * 16;
        const bf16_t* brow = vT + ((size_t)b * 256 + h * DH_ + dt * 16 + l15) * N_ + lhalf * 16;
        v8f acc = {};
        #pragma unroll 4
        for (int kk = 0; kk < 128; ++kk) {
            __builtin_prefetch(arow + kk * 32 + 512, 0, 0);   // global_prefetch_b8
            v16bf a  = make_afrag(arow + kk * 32);
            v16bf bf = *(const v16bf*)(brow + kk * 32);
            acc = __builtin_amdgcn_wmma_f32_16x16x32_bf16(false, a, false, bf,
                                                          (short)0, acc, false, false);
        }
        #pragma unroll
        for (int i = 0; i < 8; ++i) {
            const int islot = lhalf * 8 + i;
            const int r     = h * 16 + islot;
            const int dcol  = h * DH_ + dt * 16 + l15;
            const float denom = rowsum[b * 64 + r] + (float)N_ * EPS_;
            const float val = (acc[i] + EPS_ * colsum[b * 256 + dcol]) / denom;
            updates[((size_t)b * KSLOT_ + islot) * 256 + dcol] = val;
        }
    }
}

// ---------------------------------------------------------------------------
// 7) GRU cell (PyTorch gate order r,z,n), one (b,slot) row per block
__global__ __launch_bounds__(256) void gru_kernel(
    const float* __restrict__ updates, float* __restrict__ slots,
    const float* __restrict__ w_ih, const float* __restrict__ w_hh,
    const float* __restrict__ b_ih, const float* __restrict__ b_hh)
{
    __shared__ float u[DIM_], hp[DIM_];
    const int row = blockIdx.x, t = threadIdx.x;
    u[t]  = updates[(size_t)row * DIM_ + t];
    hp[t] = slots[(size_t)row * DIM_ + t];
    __syncthreads();

    float gi[3], gh[3];
    #pragma unroll
    for (int gsel = 0; gsel < 3; ++gsel) {
        const float* wi = w_ih + (size_t)(gsel * DIM_ + t) * DIM_;
        const float* wh = w_hh + (size_t)(gsel * DIM_ + t) * DIM_;
        float a = b_ih[gsel * DIM_ + t];
        float c = b_hh[gsel * DIM_ + t];
        for (int f = 0; f < DIM_; ++f) { a += u[f] * wi[f]; c += hp[f] * wh[f]; }
        gi[gsel] = a; gh[gsel] = c;
    }
    const float r = 1.0f / (1.0f + __expf(-(gi[0] + gh[0])));
    const float z = 1.0f / (1.0f + __expf(-(gi[1] + gh[1])));
    const float n = tanhf(gi[2] + r * gh[2]);
    slots[(size_t)row * DIM_ + t] = (1.0f - z) * n + z * hp[t];
}

// 8) slots += W2 @ gelu(W1 @ LN(slots) + b1) + b2
__global__ __launch_bounds__(256) void ff_kernel(
    float* __restrict__ slots, const float* __restrict__ g, const float* __restrict__ bv,
    const float* __restrict__ w1, const float* __restrict__ b1,
    const float* __restrict__ w2, const float* __restrict__ b2)
{
    __shared__ float sv[DIM_];
    __shared__ float h1[4 * DIM_];
    __shared__ float red[256];
    const int row = blockIdx.x, t = threadIdx.x;
    const float xv = slots[(size_t)row * DIM_ + t];

    red[t] = xv; __syncthreads();
    for (int st = 128; st > 0; st >>= 1) { if (t < st) red[t] += red[t + st]; __syncthreads(); }
    const float mean = red[0] * (1.0f / DIM_); __syncthreads();
    red[t] = xv * xv; __syncthreads();
    for (int st = 128; st > 0; st >>= 1) { if (t < st) red[t] += red[t + st]; __syncthreads(); }
    const float var  = red[0] * (1.0f / DIM_) - mean * mean;
    const float rstd = rsqrtf(var + 1e-5f);
    sv[t] = (xv - mean) * rstd * g[t] + bv[t];
    __syncthreads();

    #pragma unroll
    for (int rep = 0; rep < 4; ++rep) {
        const int o = t + rep * 256;
        const float* wr = w1 + (size_t)o * DIM_;
        float a = b1[o];
        for (int f = 0; f < DIM_; ++f) a += sv[f] * wr[f];
        h1[o] = 0.5f * a * (1.0f + erff(a * 0.70710678118654752f));  // exact gelu
    }
    __syncthreads();

    const float* wr2 = w2 + (size_t)t * (4 * DIM_);
    float out = b2[t];
    for (int j = 0; j < 4 * DIM_; ++j) out += h1[j] * wr2[j];
    slots[(size_t)row * DIM_ + t] = xv + out;
}

__global__ void copy_out_kernel(const float* __restrict__ slots, float* __restrict__ out) {
    int i = blockIdx.x * blockDim.x + threadIdx.x;
    if (i < B_ * KSLOT_ * DIM_) out[i] = slots[i];
}

// ---------------------------------------------------------------------------
extern "C" void kernel_launch(void* const* d_in, const int* in_sizes, int n_in,
                              void* d_out, int out_size, void* d_ws, size_t ws_size,
                              hipStream_t stream) {
    const float* inputs     = (const float*)d_in[0];
    const float* cond       = (const float*)d_in[1];
    const float* to_q_w     = (const float*)d_in[2];
    const float* to_k_w     = (const float*)d_in[3];
    const float* to_v_w     = (const float*)d_in[4];
    const float* gru_w_ih   = (const float*)d_in[5];
    const float* gru_w_hh   = (const float*)d_in[6];
    const float* gru_b_ih   = (const float*)d_in[7];
    const float* gru_b_hh   = (const float*)d_in[8];
    const float* ln_in_g    = (const float*)d_in[9];
    const float* ln_in_b    = (const float*)d_in[10];
    const float* ln_slots_g = (const float*)d_in[11];
    const float* ln_slots_b = (const float*)d_in[12];
    const float* ff_ln_g    = (const float*)d_in[13];
    const float* ff_ln_b    = (const float*)d_in[14];
    const float* ff_w1      = (const float*)d_in[15];
    const float* ff_b1      = (const float*)d_in[16];
    const float* ff_w2      = (const float*)d_in[17];
    const float* ff_b2      = (const float*)d_in[18];

    char* ws = (char*)d_ws;
    bf16_t* wk     = (bf16_t*)(ws + OFF_WK);
    bf16_t* wv     = (bf16_t*)(ws + OFF_WV);
    bf16_t* kb     = (bf16_t*)(ws + OFF_KB);
    bf16_t* vT     = (bf16_t*)(ws + OFF_VT);
    bf16_t* qb     = (bf16_t*)(ws + OFF_QB);
    bf16_t* attnW  = (bf16_t*)(ws + OFF_ATTN);
    float*  slots  = (float*)(ws + OFF_SLOTS);
    float*  upd    = (float*)(ws + OFF_UPD);
    float*  rowsum = (float*)(ws + OFF_ROWSUM);
    float*  colsum = (float*)(ws + OFF_COLSUM);

    float* out_slots = (float*)d_out;
    float* attn_out  = (float*)d_out + B_ * KSLOT_ * DIM_;

    convert_w_kernel<<<768, 256, 0, stream>>>(to_k_w, to_v_w, wk, wv);
    init_slots_kernel<<<512, 256, 0, stream>>>(cond, slots);
    ln_kv_kernel<<<(B_ * N_) / 32, 256, 0, stream>>>(inputs, ln_in_g, ln_in_b,
                                                     wk, wv, kb, vT);
    colsum_kernel<<<B_ * 256, 256, 0, stream>>>(vT, colsum);

    for (int it = 0; it < ITERS_; ++it) {
        zero_kernel<<<8, 256, 0, stream>>>(rowsum, B_ * 64);
        qproj_kernel<<<B_ * KSLOT_, 256, 0, stream>>>(slots, ln_slots_g, ln_slots_b,
                                                      to_q_w, qb);
        dots_softmax_kernel<<<B_ * 64, 256, 0, stream>>>(qb, kb, attnW, rowsum,
                                                         attn_out, (it == ITERS_ - 1) ? 1 : 0);
        updates_kernel<<<B_ * H_, 64, 0, stream>>>(attnW, vT, rowsum, colsum, upd);
        gru_kernel<<<B_ * KSLOT_, 256, 0, stream>>>(upd, slots, gru_w_ih, gru_w_hh,
                                                    gru_b_ih, gru_b_hh);
        ff_kernel<<<B_ * KSLOT_, 256, 0, stream>>>(slots, ff_ln_g, ff_ln_b,
                                                   ff_w1, ff_b1, ff_w2, ff_b2);
    }
    copy_out_kernel<<<512, 256, 0, stream>>>(slots, out_slots);
}
